// LSTM_77051713290896
// MI455X (gfx1250) — matrix operand order repaired and checked by reading, not compiled, not run on hardware
//
#include <hip/hip_runtime.h>
#include <hip/hip_bf16.h>

typedef __bf16 bf16_t;
typedef __attribute__((ext_vector_type(16))) __bf16 v16bf;
typedef __attribute__((ext_vector_type(8)))  __bf16 v8bf;
typedef __attribute__((ext_vector_type(8)))  float  v8f;

#define BSZ   128
#define TLEN  256
#define HH    256
#define GATES 1024
#define GPAD  1028   // fp32 gate buffer row stride (+4 dwords kills bank conflicts)

__device__ __forceinline__ float sigmoidf_(float x) {
    return 1.0f / (1.0f + __expf(-x));
}

// ---------------------------------------------------------------------------
// Pack [W_ih | W_hh] (fp32, row-major [4H][K_in]/[4H][H]) into one bf16 matrix
// W[n][k], k in [0, K_total), per direction. Also fuse b_ih + b_hh (fp32).
// ---------------------------------------------------------------------------
__global__ void pack_weights_kernel(const float* __restrict__ wih_f, const float* __restrict__ whh_f,
                                    const float* __restrict__ bih_f, const float* __restrict__ bhh_f,
                                    const float* __restrict__ wih_b, const float* __restrict__ whh_b,
                                    const float* __restrict__ bih_b, const float* __restrict__ bhh_b,
                                    bf16_t* __restrict__ Wpack, float* __restrict__ bias,
                                    int K_in, int K_total)
{
    int idx = blockIdx.x * blockDim.x + threadIdx.x;
    int total = 2 * GATES * K_total;
    if (idx < total) {
        int dir = idx / (GATES * K_total);
        int rem = idx % (GATES * K_total);
        int n = rem / K_total;
        int k = rem % K_total;
        const float* wih = dir ? wih_b : wih_f;
        const float* whh = dir ? whh_b : whh_f;
        float v = (k < K_in) ? wih[n * K_in + k] : whh[n * HH + (k - K_in)];
        Wpack[idx] = (bf16_t)v;
    }
    if (idx < 2 * GATES) {
        int dir = idx / GATES;
        int n   = idx % GATES;
        bias[idx] = dir ? (bih_b[n] + bhh_b[n]) : (bih_f[n] + bhh_f[n]);
    }
}

// ---------------------------------------------------------------------------
// Persistent bidirectional LSTM layer.
// grid = 16 blocks: dir = blockIdx.x>>3, batch slice m0 = (blockIdx.x&7)*16.
// block = 256 threads = 8 waves; each wave owns a 128-wide gate slice.
// Per step: gates[16x1024] = [x_t | h_prev][16 x K_total] * Wpack^T  via
// v_wmma_f32_16x16x32_bf16, then fp32 gate nonlinearities + c/h update.
// ---------------------------------------------------------------------------
__global__ void __launch_bounds__(256)
lstm_layer_kernel(const float*  __restrict__ xin_f32,   // layer-0 input (or null)
                  const bf16_t* __restrict__ xin_bf,    // layer-1 input (or null)
                  const bf16_t* __restrict__ Wpack,     // [2][GATES*K_total] bf16
                  const float*  __restrict__ bias,      // [2][GATES] fp32
                  bf16_t* __restrict__ out_all,         // [B][T][2H] bf16, or null
                  float*  __restrict__ out_last,        // [B][2H] fp32, or null
                  int K_in, int K_total, int K_pad)
{
    extern __shared__ __align__(16) char smem_raw[];
    bf16_t* A_lds = (bf16_t*)smem_raw;                                   // [16][K_pad]
    float*  gbuf  = (float*)(smem_raw + (size_t)16 * K_pad * sizeof(bf16_t)); // [16][GPAD]

    const int tid  = threadIdx.x;
    const int lane = tid & 31;
    const int wv   = tid >> 5;                 // wave 0..7
    const int dir  = blockIdx.x >> 3;          // 0 = fwd, 1 = bwd
    const int m0   = (blockIdx.x & 7) << 4;    // batch slice start

    const bf16_t* Wd = Wpack + (size_t)dir * GATES * K_total;
    const float*  bd = bias + dir * GATES;

    // h_prev = 0 in A region; c = 0 in registers
    for (int idx = tid; idx < 16 * HH; idx += 256) {
        int e = idx >> 8, j = idx & 255;
        A_lds[e * K_pad + K_in + j] = (bf16_t)0.0f;
    }
    float creg[16];
#pragma unroll
    for (int e = 0; e < 16; ++e) creg[e] = 0.0f;
    __syncthreads();

    // WMMA fragment lane geometry (16-bit A 16x32 / B 32x16 layouts)
    const int am  = lane & 15;          // A: row M
    const int akh = (lane >> 4) * 8;    // A: K-half offset {0,8}
    const int bn  = lane & 15;          // B/C/D: column within tile
    const int bkh = (lane >> 4) * 8;    // B: K-half offset {0,8}
    const int gm  = (lane >> 4) * 8;    // C/D: row offset {0,8}
    const int n0  = wv * 128;           // this wave's gate-column base

    for (int t = 0; t < TLEN; ++t) {
        const int t_real = dir ? (TLEN - 1 - t) : t;

        // ---- stage x_t into A[:, 0:K_in] (bf16) ----
        if (xin_f32) {
            for (int idx = tid; idx < 16 * K_in; idx += 256) {
                int e = idx / K_in, k = idx % K_in;
                A_lds[e * K_pad + k] =
                    (bf16_t)xin_f32[((size_t)(m0 + e) * TLEN + t_real) * K_in + k];
            }
        } else {
            const int vpr = K_in >> 3;            // uint4 vectors per row
            for (int idx = tid; idx < 16 * vpr; idx += 256) {
                int e = idx / vpr, kv = idx % vpr;
                *(uint4*)(A_lds + e * K_pad + kv * 8) =
                    *(const uint4*)(xin_bf + ((size_t)(m0 + e) * TLEN + t_real) * K_in + kv * 8);
            }
        }
        __syncthreads();

        // ---- GEMM: gates = A * W^T (bf16 WMMA, fp32 accumulate) ----
        v8f acc[8];
#pragma unroll
        for (int tt = 0; tt < 8; ++tt)
#pragma unroll
            for (int i = 0; i < 8; ++i) acc[tt][i] = 0.0f;

        for (int kc = 0; kc < K_total; kc += 32) {
            v8bf alo = *(const v8bf*)(A_lds + am * K_pad + kc + akh);
            v8bf ahi = *(const v8bf*)(A_lds + am * K_pad + kc + 16 + akh);
            v16bf afrag;
#pragma unroll
            for (int i = 0; i < 8; ++i) { afrag[i] = alo[i]; afrag[i + 8] = ahi[i]; }

#pragma unroll
            for (int tt = 0; tt < 8; ++tt) {
                const bf16_t* wp = Wd + (size_t)(n0 + tt * 16 + bn) * K_total + kc + bkh;
                v8bf blo = *(const v8bf*)(wp);
                v8bf bhi = *(const v8bf*)(wp + 16);
                v16bf bfrag;
#pragma unroll
                for (int i = 0; i < 8; ++i) { bfrag[i] = blo[i]; bfrag[i + 8] = bhi[i]; }
                acc[tt] = __builtin_amdgcn_wmma_f32_16x16x32_bf16(
                    false, afrag, false, bfrag, (short)0, acc[tt], false, false);
            }
        }

        // ---- scatter gate tiles to LDS (fp32) ----
#pragma unroll
        for (int tt = 0; tt < 8; ++tt) {
#pragma unroll
            for (int r = 0; r < 8; ++r) {
                gbuf[(gm + r) * GPAD + n0 + tt * 16 + bn] = acc[tt][r];
            }
        }
        __syncthreads();

        // ---- elementwise: thread tid owns hidden column j = tid, rows 0..15 ----
        {
            const int j = tid;
            const float bi = bd[j], bf_ = bd[256 + j], bg = bd[512 + j], bo = bd[768 + j];
#pragma unroll
            for (int e = 0; e < 16; ++e) {
                float ig = sigmoidf_(gbuf[e * GPAD + j]       + bi);
                float fg = sigmoidf_(gbuf[e * GPAD + 256 + j] + bf_);
                float gg = tanhf(    gbuf[e * GPAD + 512 + j] + bg);
                float og = sigmoidf_(gbuf[e * GPAD + 768 + j] + bo);
                float c = fg * creg[e] + ig * gg;
                creg[e] = c;
                float h = og * tanhf(c);
                A_lds[e * K_pad + K_in + j] = (bf16_t)h;       // h_prev for next step
                if (out_all) {
                    out_all[((size_t)(m0 + e) * TLEN + t_real) * (2 * HH) + dir * HH + j] = (bf16_t)h;
                }
                if (out_last && t_real == TLEN - 1) {
                    out_last[(size_t)(m0 + e) * (2 * HH) + dir * HH + j] = h;
                }
            }
        }
        __syncthreads();
    }
}

// ---------------------------------------------------------------------------
// Final FC: out[b] = last[b,:] . fc_w[0,:] + fc_b[0]   (NC == 1)
// ---------------------------------------------------------------------------
__global__ void fc_kernel(const float* __restrict__ last, const float* __restrict__ fc_w,
                          const float* __restrict__ fc_b, float* __restrict__ out)
{
    int b = blockIdx.x * blockDim.x + threadIdx.x;
    if (b < BSZ) {
        float s = fc_b[0];
#pragma unroll 8
        for (int j = 0; j < 2 * HH; ++j)
            s += last[(size_t)b * 2 * HH + j] * fc_w[j];
        out[b] = s;
    }
}

extern "C" void kernel_launch(void* const* d_in, const int* in_sizes, int n_in,
                              void* d_out, int out_size, void* d_ws, size_t ws_size,
                              hipStream_t stream)
{
    (void)in_sizes; (void)n_in; (void)out_size; (void)ws_size;
    const float* x    = (const float*)d_in[0];
    const float* fc_w = (const float*)d_in[17];
    const float* fc_b = (const float*)d_in[18];

    size_t off = 0;
    auto carve = [&](size_t bytes) -> void* {
        void* p = (char*)d_ws + off;
        off += (bytes + 255) & ~(size_t)255;
        return p;
    };
    bf16_t* Wp0   = (bf16_t*)carve((size_t)2 * GATES * 320 * sizeof(bf16_t));
    bf16_t* Wp1   = (bf16_t*)carve((size_t)2 * GATES * 768 * sizeof(bf16_t));
    float*  b0    = (float*) carve((size_t)2 * GATES * sizeof(float));
    float*  b1    = (float*) carve((size_t)2 * GATES * sizeof(float));
    bf16_t* out0  = (bf16_t*)carve((size_t)BSZ * TLEN * 2 * HH * sizeof(bf16_t));
    float*  last1 = (float*) carve((size_t)BSZ * 2 * HH * sizeof(float));

    {   // pack layer 0 weights (K_in=64, K_total=320)
        int total = 2 * GATES * 320;
        pack_weights_kernel<<<(total + 255) / 256, 256, 0, stream>>>(
            (const float*)d_in[1], (const float*)d_in[2], (const float*)d_in[3], (const float*)d_in[4],
            (const float*)d_in[5], (const float*)d_in[6], (const float*)d_in[7], (const float*)d_in[8],
            Wp0, b0, 64, 320);
    }
    {   // pack layer 1 weights (K_in=512, K_total=768)
        int total = 2 * GATES * 768;
        pack_weights_kernel<<<(total + 255) / 256, 256, 0, stream>>>(
            (const float*)d_in[9],  (const float*)d_in[10], (const float*)d_in[11], (const float*)d_in[12],
            (const float*)d_in[13], (const float*)d_in[14], (const float*)d_in[15], (const float*)d_in[16],
            Wp1, b1, 512, 768);
    }

    {   // layer 0: fp32 x input, store full bf16 output
        const int K_pad = 328;
        size_t shmem = (size_t)16 * K_pad * sizeof(bf16_t) + (size_t)16 * GPAD * sizeof(float);
        lstm_layer_kernel<<<16, 256, shmem, stream>>>(
            x, nullptr, Wp0, b0, out0, nullptr, 64, 320, K_pad);
    }
    {   // layer 1: bf16 input from layer 0, store only last-timestep fp32
        const int K_pad = 776;
        size_t shmem = (size_t)16 * K_pad * sizeof(bf16_t) + (size_t)16 * GPAD * sizeof(float);
        lstm_layer_kernel<<<16, 256, shmem, stream>>>(
            nullptr, out0, Wp1, b1, nullptr, last1, 512, 768, K_pad);
    }

    fc_kernel<<<1, 128, 0, stream>>>(last1, fc_w, fc_b, (float*)d_out);
}
